// PCB_Effi_GGNN_8211977470089
// MI455X (gfx1250) — compile-verified, matrix operand order checked by reading, not run on hardware
//
#include <hip/hip_runtime.h>
#include <math.h>

typedef __bf16 bf16_t;
typedef __bf16 bf16x8 __attribute__((ext_vector_type(8)));
typedef __bf16 v16bf  __attribute__((ext_vector_type(16)));
typedef float  v8f    __attribute__((ext_vector_type(8)));
typedef float  f32x4  __attribute__((ext_vector_type(4)));

#define DD     1280
#define NNODE  6
#define NEDGE  6
#define BBATCH 1024
#define MROWS  (BBATCH*NNODE)   // 6144
#define KEDG   (NEDGE*DD)       // 7680
#define K3     (3*DD)           // 3840
#define N2     (2*DD)           // 2560

// ---------------- weight conversion (once per launch) ----------------

// W (E,D,D) f32 -> WT (D, E*D) bf16 : WT[f*KE + e*D + d] = W[e,d,f]
__global__ void cvt_w_edge(const float* __restrict__ W, bf16_t* __restrict__ WT) {
  int tid = blockIdx.x * blockDim.x + threadIdx.x;
  if (tid >= DD * KEDG) return;
  int f = tid / KEDG;
  int c = tid % KEDG;
  int e = c / DD, d = c % DD;
  WT[tid] = (bf16_t)W[(size_t)e * DD * DD + (size_t)d * DD + f];
}

// W_r,W_z (3D,D) f32 -> WT (2D, 3D) bf16 (r rows then z rows, transposed)
__global__ void cvt_w_rz(const float* __restrict__ Wr, const float* __restrict__ Wz,
                         bf16_t* __restrict__ WT) {
  int tid = blockIdx.x * blockDim.x + threadIdx.x;
  if (tid >= N2 * K3) return;
  int f2 = tid / K3;
  int k  = tid % K3;
  float v = (f2 < DD) ? Wr[(size_t)k * DD + f2] : Wz[(size_t)k * DD + (f2 - DD)];
  WT[tid] = (bf16_t)v;
}

// W_t (3D,D) f32 -> WT (D, 3D) bf16
__global__ void cvt_w_t(const float* __restrict__ Wt, bf16_t* __restrict__ WT) {
  int tid = blockIdx.x * blockDim.x + threadIdx.x;
  if (tid >= DD * K3) return;
  int f = tid / K3;
  int k = tid % K3;
  WT[tid] = (bf16_t)Wt[(size_t)k * DD + f];
}

// effective bias: bias_eff[n,f] = sum_e (sum_m A[n, half*36 + e*6 + m]) * b[e,f]
__global__ void bias_eff_kernel(const float* __restrict__ Aadj,
                                const float* __restrict__ b_in,
                                const float* __restrict__ b_out,
                                float* __restrict__ bias_in,
                                float* __restrict__ bias_out) {
  int tid = blockIdx.x * blockDim.x + threadIdx.x;
  if (tid >= 2 * NNODE * DD) return;
  int half = tid / (NNODE * DD);
  int r = tid % (NNODE * DD);
  int n = r / DD, f = r % DD;
  const float* bb = half ? b_out : b_in;
  float acc = 0.f;
  for (int e = 0; e < NEDGE; e++) {
    float rs = 0.f;
    for (int m = 0; m < NNODE; m++)
      rs += Aadj[n * (NNODE * NEDGE * 2) + half * 36 + e * NNODE + m];
    acc += rs * bb[e * DD + f];
  }
  (half ? bias_out : bias_in)[r] = acc;
}

// ---------------- per-step prep ----------------

// adjacency pre-aggregation: G[(b,n), e*D+d] = sum_m A[n, aoff + e*6+m] * gx[b,m,d]  (bf16 out)
__global__ void prep_G(const float* __restrict__ gx, const float* __restrict__ Aadj,
                       int aoff, bf16_t* __restrict__ G) {
  int i8 = blockIdx.x * blockDim.x + threadIdx.x;
  const int cols8 = KEDG / 8;  // 960
  if (i8 >= MROWS * cols8) return;
  int row = i8 / cols8;
  int c8  = i8 % cols8;
  int e   = c8 / (DD / 8);
  int d8  = c8 % (DD / 8);
  int n = row % NNODE;
  int b = row / NNODE;
  float acc[8];
#pragma unroll
  for (int i = 0; i < 8; i++) acc[i] = 0.f;
  const float* arow = Aadj + n * (NNODE * NEDGE * 2) + aoff + e * NNODE;
#pragma unroll
  for (int m = 0; m < NNODE; m++) {
    float cf = arow[m];
    if (cf != 0.0f) {
      const float* g = gx + (size_t)(b * NNODE + m) * DD + d8 * 8;
      f32x4 lo = *reinterpret_cast<const f32x4*>(g);
      f32x4 hh = *reinterpret_cast<const f32x4*>(g + 4);
#pragma unroll
      for (int i = 0; i < 4; i++) { acc[i] += cf * lo[i]; acc[4 + i] += cf * hh[i]; }
    }
  }
  bf16x8 o;
#pragma unroll
  for (int i = 0; i < 8; i++) o[i] = (bf16_t)acc[i];
  *reinterpret_cast<bf16x8*>(G + (size_t)row * KEDG + c8 * 8) = o;
}

// write bf16 copy of gx into Acat columns [2560, 3840)
__global__ void gx_to_bf16(const float* __restrict__ gx, bf16_t* __restrict__ Acat) {
  int i8 = blockIdx.x * blockDim.x + threadIdx.x;
  const int cols8 = DD / 8;  // 160
  if (i8 >= MROWS * cols8) return;
  int row = i8 / cols8, d8 = i8 % cols8;
  const float* g = gx + (size_t)row * DD + d8 * 8;
  f32x4 lo = *reinterpret_cast<const f32x4*>(g);
  f32x4 hh = *reinterpret_cast<const f32x4*>(g + 4);
  bf16x8 o;
#pragma unroll
  for (int i = 0; i < 4; i++) { o[i] = (bf16_t)lo[i]; o[4 + i] = (bf16_t)hh[i]; }
  *reinterpret_cast<bf16x8*>(Acat + (size_t)row * K3 + 2 * DD + d8 * 8) = o;
}

// ---------------- WMMA GEMM (bf16 in, f32 acc) ----------------
// C(row, col) = sum_k A(row, k) * BT(col, k)
// A comes from segment A0 (k < kSplit, stride ldA0) or A1 (k >= kSplit, stride ldA1).
// MODE 0: a_in/a_out -> +bias_eff[(row%6), col], store bf16 to outB (ld outB_ld)
// MODE 1: r/z        -> col<D: r=sig(v+b_r[col]), outB[row*D+col]=bf16(r*gx)
//                       col>=D: z=sig(v+b_z[f]),  outF[row*D+f]=z
// MODE 2: h_hat      -> h=tanh(v+b_t[col]); outF = (1-z)*gx + z*h
template <int MODE>
__global__ __launch_bounds__(256) void gemm_wmma(
    const bf16_t* __restrict__ A0, int ldA0,
    const bf16_t* __restrict__ A1, int ldA1, int kSplit,
    const bf16_t* __restrict__ BT, int K,
    float* __restrict__ outF, bf16_t* __restrict__ outB, int outB_ld,
    const float* __restrict__ bias0, const float* __restrict__ bias1,
    const float* __restrict__ gx, const float* __restrict__ zbuf) {
  const int wave = threadIdx.x >> 5;
  const int lane = threadIdx.x & 31;
  const int wm = wave & 1;   // 0..1 -> M
  const int wn = wave >> 1;  // 0..3 -> N
  const int m0 = blockIdx.x * 64 + wm * 32;
  const int n0 = blockIdx.y * 256 + wn * 64;
  const int lm = lane & 15;
  const int hf = lane >> 4;      // 0/1
  const int koff = hf * 8;

  const v8f vzero = {0.f, 0.f, 0.f, 0.f, 0.f, 0.f, 0.f, 0.f};
  v8f acc[2][4];
#pragma unroll
  for (int mt = 0; mt < 2; mt++)
#pragma unroll
    for (int nt = 0; nt < 4; nt++) acc[mt][nt] = vzero;

  for (int k = 0; k < K; k += 32) {
    const bf16_t* Ab;
    int ld, kk;
    if (k < kSplit) { Ab = A0; ld = ldA0; kk = k; }
    else            { Ab = A1; ld = ldA1; kk = k - kSplit; }

    v16bf afr[2];
#pragma unroll
    for (int mt = 0; mt < 2; mt++) {
      const bf16_t* p = Ab + (size_t)(m0 + mt * 16 + lm) * ld + kk + koff;
      bf16x8 lo = *reinterpret_cast<const bf16x8*>(p);
      bf16x8 hh = *reinterpret_cast<const bf16x8*>(p + 16);
      afr[mt] = __builtin_shufflevector(lo, hh, 0, 1, 2, 3, 4, 5, 6, 7,
                                        8, 9, 10, 11, 12, 13, 14, 15);
    }
#pragma unroll
    for (int nt = 0; nt < 4; nt++) {
      const bf16_t* p = BT + (size_t)(n0 + nt * 16 + lm) * K + k + koff;
      bf16x8 lo = *reinterpret_cast<const bf16x8*>(p);
      bf16x8 hh = *reinterpret_cast<const bf16x8*>(p + 16);
      v16bf bfr = __builtin_shufflevector(lo, hh, 0, 1, 2, 3, 4, 5, 6, 7,
                                          8, 9, 10, 11, 12, 13, 14, 15);
#pragma unroll
      for (int mt = 0; mt < 2; mt++)
        acc[mt][nt] = __builtin_amdgcn_wmma_f32_16x16x32_bf16(
            false, afr[mt], false, bfr, (short)0, acc[mt][nt], false, false);
    }
  }

#pragma unroll
  for (int mt = 0; mt < 2; mt++) {
#pragma unroll
    for (int nt = 0; nt < 4; nt++) {
      const int colb = n0 + nt * 16 + lm;
#pragma unroll
      for (int j = 0; j < 8; j++) {
        const int row = m0 + mt * 16 + hf * 8 + j;
        float val = acc[mt][nt][j];
        if (MODE == 0) {
          val += bias0[(row % NNODE) * DD + colb];
          outB[(size_t)row * outB_ld + colb] = (bf16_t)val;
        } else if (MODE == 1) {
          if (colb < DD) {
            float r = 1.f / (1.f + __expf(-(val + bias0[colb])));
            outB[(size_t)row * DD + colb] = (bf16_t)(r * gx[(size_t)row * DD + colb]);
          } else {
            int f = colb - DD;
            float z = 1.f / (1.f + __expf(-(val + bias1[f])));
            outF[(size_t)row * DD + f] = z;
          }
        } else {
          float h = tanhf(val + bias0[colb]);
          float z = zbuf[(size_t)row * DD + colb];
          float g = gx[(size_t)row * DD + colb];
          outF[(size_t)row * DD + colb] = (1.f - z) * g + z * h;
        }
      }
    }
  }
}

// out[b, d*6+n] = gx[b, n, d]
__global__ void final_transpose(const float* __restrict__ gx, float* __restrict__ out) {
  int tid = blockIdx.x * blockDim.x + threadIdx.x;
  if (tid >= BBATCH * NNODE * DD) return;
  int b = tid / (NNODE * DD);
  int r = tid % (NNODE * DD);
  int d = r / NNODE;
  int n = r % NNODE;
  out[tid] = gx[((size_t)b * NNODE + n) * DD + d];
}

// ---------------- launcher ----------------

extern "C" void kernel_launch(void* const* d_in, const int* in_sizes, int n_in,
                              void* d_out, int out_size, void* d_ws, size_t ws_size,
                              hipStream_t stream) {
  const float* x    = (const float*)d_in[0];
  const float* A    = (const float*)d_in[1];
  const float* W_in = (const float*)d_in[2];
  const float* b_in = (const float*)d_in[3];
  const float* W_out= (const float*)d_in[4];
  const float* b_out= (const float*)d_in[5];
  const float* W_r  = (const float*)d_in[6];
  const float* b_r  = (const float*)d_in[7];
  const float* W_z  = (const float*)d_in[8];
  const float* b_z  = (const float*)d_in[9];
  const float* W_t  = (const float*)d_in[10];
  const float* b_t  = (const float*)d_in[11];
  float* out = (float*)d_out;

  char* base = (char*)d_ws;
  size_t off = 0;
  auto carve = [&](size_t bytes) -> char* {
    char* p = base + off;
    off += (bytes + 255) & ~(size_t)255;
    return p;
  };
  float*  gx0    = (float*)carve((size_t)MROWS * DD * 4);
  float*  gx1    = (float*)carve((size_t)MROWS * DD * 4);
  bf16_t* G      = (bf16_t*)carve((size_t)MROWS * KEDG * 2);
  bf16_t* Acat   = (bf16_t*)carve((size_t)MROWS * K3 * 2);
  bf16_t* rgx    = (bf16_t*)carve((size_t)MROWS * DD * 2);
  float*  zbuf   = (float*)carve((size_t)MROWS * DD * 4);
  bf16_t* WinT   = (bf16_t*)carve((size_t)DD * KEDG * 2);
  bf16_t* WoutT  = (bf16_t*)carve((size_t)DD * KEDG * 2);
  bf16_t* WrzT   = (bf16_t*)carve((size_t)N2 * K3 * 2);
  bf16_t* WtT    = (bf16_t*)carve((size_t)DD * K3 * 2);
  float*  biasin = (float*)carve((size_t)NNODE * DD * 4);
  float*  biasout= (float*)carve((size_t)NNODE * DD * 4);

  const int THR = 256;
  cvt_w_edge<<<(DD * KEDG + THR - 1) / THR, THR, 0, stream>>>(W_in, WinT);
  cvt_w_edge<<<(DD * KEDG + THR - 1) / THR, THR, 0, stream>>>(W_out, WoutT);
  cvt_w_rz<<<(N2 * K3 + THR - 1) / THR, THR, 0, stream>>>(W_r, W_z, WrzT);
  cvt_w_t<<<(DD * K3 + THR - 1) / THR, THR, 0, stream>>>(W_t, WtT);
  bias_eff_kernel<<<(2 * NNODE * DD + THR - 1) / THR, THR, 0, stream>>>(
      A, b_in, b_out, biasin, biasout);

  const int BIG = 1 << 30;
  for (int step = 0; step < 5; step++) {
    const float* src = (step == 0) ? x : ((step & 1) ? gx0 : gx1);
    float* dst = (step & 1) ? gx1 : gx0;

    gx_to_bf16<<<(MROWS * (DD / 8) + THR - 1) / THR, THR, 0, stream>>>(src, Acat);

    // a_in
    prep_G<<<(MROWS * (KEDG / 8) + THR - 1) / THR, THR, 0, stream>>>(src, A, 0, G);
    gemm_wmma<0><<<dim3(MROWS / 64, DD / 256), THR, 0, stream>>>(
        G, KEDG, G, KEDG, BIG, WinT, KEDG,
        nullptr, Acat + 0, K3, biasin, nullptr, nullptr, nullptr);

    // a_out (G buffer reused; stream order serializes)
    prep_G<<<(MROWS * (KEDG / 8) + THR - 1) / THR, THR, 0, stream>>>(src, A, 36, G);
    gemm_wmma<0><<<dim3(MROWS / 64, DD / 256), THR, 0, stream>>>(
        G, KEDG, G, KEDG, BIG, WoutT, KEDG,
        nullptr, Acat + DD, K3, biasout, nullptr, nullptr, nullptr);

    // r,z gates: one GEMM over [a_in|a_out|gx] with [W_r|W_z]
    gemm_wmma<1><<<dim3(MROWS / 64, N2 / 256), THR, 0, stream>>>(
        Acat, K3, Acat, K3, BIG, WrzT, K3,
        zbuf, rgx, DD, b_r, b_z, src, nullptr);

    // h_hat + state update: A-segments [a_in|a_out] then r*gx
    gemm_wmma<2><<<dim3(MROWS / 64, DD / 256), THR, 0, stream>>>(
        Acat, K3, rgx, DD, 2 * DD, WtT, K3,
        dst, nullptr, 0, b_t, nullptr, src, zbuf);
  }

  final_transpose<<<(BBATCH * NNODE * DD + THR - 1) / THR, THR, 0, stream>>>(gx0, out);
}
